// State_Encoding_86517821216365
// MI455X (gfx1250) — compile-verified, matrix operand order checked by reading, not run on hardware
//
#include <hip/hip_runtime.h>
#include <hip/hip_bf16.h>

#define BB  16
#define TT  2048
#define RR  256
#define NSS 4
#define VLD 40    // padded row stride (u16) of V tile rows: 80B, 16B-aligned, bank stride 20
#define SKLD 264  // padded row stride (u16) of K tile / staged W: 528B, bank stride 4
#define PLD 40    // padded row stride (u16) of P tile

typedef __attribute__((ext_vector_type(16))) __bf16 v16bf;
typedef __attribute__((ext_vector_type(2)))  __bf16 v2bf;
typedef __attribute__((ext_vector_type(8)))  float  v8f;

union AFrag { v16bf v; unsigned int d[8]; };

__device__ __forceinline__ unsigned short bf(float x){
  return __builtin_bit_cast(unsigned short, (__bf16)x);
}
__device__ __forceinline__ unsigned int packbf(float a, float b){
  v2bf p; p[0] = (__bf16)a; p[1] = (__bf16)b;
  return __builtin_bit_cast(unsigned int, p);
}

// 16x32 bf16 fragment (ISA 7.12.2 layout): per lane two contiguous 16B runs at
// row*(ld) + hf*8 and +16 elements -> forces b128 loads. Row index = lane&15.
__device__ __forceinline__ AFrag load_frag_bf16(const unsigned short* base, int ld, int lane){
  const int r = lane & 15, hf = lane >> 4;
  const unsigned short* p = base + r*ld + hf*8;
  const uint4 lo = *(const uint4*)(p);
  const uint4 hi = *(const uint4*)(p + 16);
  AFrag f;
  f.d[0]=lo.x; f.d[1]=lo.y; f.d[2]=lo.z; f.d[3]=lo.w;
  f.d[4]=hi.x; f.d[5]=hi.y; f.d[6]=hi.z; f.d[7]=hi.w;
  return f;
}
__device__ __forceinline__ v8f wmma_bf16(AFrag a, AFrag b, v8f c){
  return __builtin_amdgcn_wmma_f32_16x16x32_bf16(false, a.v, false, b.v, (short)0, c, false, false);
}

// ---------------------------------------------------------------------------
// Kernel 0: streaming f32 -> bf16 conversion (memory-bound)
// ---------------------------------------------------------------------------
__global__ __launch_bounds__(256) void cvt_kernel(const float* __restrict__ src,
                                                  unsigned short* __restrict__ dst, int n4){
  const int stride = gridDim.x * 256;
  for (int i = blockIdx.x*256 + threadIdx.x; i < n4; i += stride){
    const float4 x = ((const float4*)src)[i];
    uint2 o; o.x = packbf(x.x, x.y); o.y = packbf(x.z, x.w);
    ((uint2*)dst)[i] = o;
  }
}

// ---------------------------------------------------------------------------
// Kernel 1: fused projections. grid = (B*T/128, 6); sel 0=qp, 1=kp, 2..5=vs[n]
// W staged once per block into padded LDS (shared by 8 waves). vs is written
// TRANSPOSED [B,NS,R,T] with one b128 store per lane per tile (L2 merges).
// ---------------------------------------------------------------------------
__global__ __launch_bounds__(256) void proj_kernel(
    const unsigned short* __restrict__ qb, const unsigned short* __restrict__ kb,
    const unsigned short* __restrict__ vb,
    const unsigned short* __restrict__ wqb, const float* __restrict__ bq,
    const unsigned short* __restrict__ wkb, const float* __restrict__ bk,
    const unsigned short* __restrict__ wvb, const float* __restrict__ bv,
    const float* __restrict__ wp, const float* __restrict__ bp,
    unsigned short* __restrict__ qp, unsigned short* __restrict__ kp,
    unsigned short* __restrict__ vst, float* __restrict__ proj)
{
  extern __shared__ unsigned short sW[];            // 256 x SKLD (132 KB)
  const int sel  = blockIdx.y;
  const int tid  = threadIdx.x;
  const int lane = tid & 31, wave = tid >> 5;
  const int ln   = lane & 15, hf = lane >> 4;
  const int g0   = blockIdx.x * 128 + wave * 16;    // first flattened (b,t) row

  const unsigned short *src, *W; const float* bias;
  if (sel == 0)      { src = qb; W = wqb; bias = bq; }
  else if (sel == 1) { src = kb; W = wkb; bias = bk; }
  else { const int n = sel - 2; src = vb; W = wvb + (size_t)n*RR*RR; bias = bv + n*RR; }

  // A fragments from global (per-wave rows), issued before W staging
  AFrag aq[8];
#pragma unroll
  for (int kk = 0; kk < 8; ++kk)
    aq[kk] = load_frag_bf16(src + (size_t)g0*RR + kk*32, RR, lane);

  // cooperative stage of W into LDS with padded rows (512B data + 16B pad)
  {
    const uint4* gw = (const uint4*)W;
    for (int i = tid; i < 8192; i += 256){
      const int r = i >> 5, j = i & 31;
      *(uint4*)(sW + r*SKLD + j*8) = gw[i];
    }
  }
  __syncthreads();

  v8f acc[16];
#pragma unroll
  for (int t = 0; t < 16; ++t){
    const float bbias = bias[t*16 + ln];            // C layout: col = ln
#pragma unroll
    for (int e = 0; e < 8; ++e) acc[t][e] = bbias;
  }
#pragma unroll
  for (int t = 0; t < 16; ++t){
    AFrag bfr[8];
#pragma unroll
    for (int kk = 0; kk < 8; ++kk)                  // batched DS issue (b128)
      bfr[kk] = load_frag_bf16(sW + (t*16)*SKLD + kk*32, SKLD, lane);
#pragma unroll
    for (int kk = 0; kk < 8; ++kk)
      acc[t] = wmma_bf16(aq[kk], bfr[kk], acc[t]);
  }
  if (sel < 2){
    unsigned short* dst = sel ? kp : qp;
#pragma unroll
    for (int e = 0; e < 8; ++e){
      const size_t row = (size_t)(g0 + e + 8*hf);
#pragma unroll
      for (int t = 0; t < 16; ++t)
        dst[row*RR + t*16 + ln] = bf(acc[t][e]);
    }
  } else {
    const int n = sel - 2;
    const int b2  = g0 >> 11;                       // whole block in one batch
    const int tl0 = (g0 & (TT-1)) + 8*hf;           // s-index of e=0
    float ps[8];
#pragma unroll
    for (int e = 0; e < 8; ++e) ps[e] = 0.f;
#pragma unroll
    for (int t = 0; t < 16; ++t){
      // pack 8 consecutive-s bf16 values into one b128 store: vst[b,n,r,s0..s0+7]
      uint4 w;
      w.x = packbf(acc[t][0], acc[t][1]);
      w.y = packbf(acc[t][2], acc[t][3]);
      w.z = packbf(acc[t][4], acc[t][5]);
      w.w = packbf(acc[t][6], acc[t][7]);
      const size_t rrow = ((size_t)b2*NSS + n)*RR + (t*16 + ln);
      *(uint4*)(vst + rrow*TT + tl0) = w;
      const float wpc = wp[t*16 + ln];
#pragma unroll
      for (int e = 0; e < 8; ++e) ps[e] += acc[t][e] * wpc;
    }
#pragma unroll
    for (int e = 0; e < 8; ++e){
      float p = ps[e];
      p += __shfl_xor(p, 1, 32); p += __shfl_xor(p, 2, 32);
      p += __shfl_xor(p, 4, 32); p += __shfl_xor(p, 8, 32);
      if (ln == 0) proj[((size_t)b2*NSS + n)*TT + tl0 + e] = p + bp[0];
    }
  }
}

// ---------------------------------------------------------------------------
// Kernel 2: causal flash attention + AV, grid = (B*T/128, NS).
// Double-buffered LDS; BOTH K and V staged via async-to-LDS (ASYNCcnt).
// One barrier per chunk.
// ---------------------------------------------------------------------------
#define SKELEM (32*SKLD)   // one K buffer, u16 elems (padded rows)
#define SVELEM (RR*VLD)    // one V buffer, u16 elems (row r -> 32 s values)
#define SPELEM (16*PLD)    // one per-wave P tile, u16 elems

__global__ __launch_bounds__(256) void attn_kernel(
    const unsigned short* __restrict__ qp, const unsigned short* __restrict__ kp,
    const unsigned short* __restrict__ vst, float* __restrict__ out)
{
  extern __shared__ unsigned short smem[];
  unsigned short* s_kb  = smem;                       // 2 x 32xSKLD, [s][r]
  unsigned short* s_vtb = smem + 2*SKELEM;            // 2 x RRxVLD,  [r][s]
  unsigned short* s_p   = smem + 2*SKELEM + 2*SVELEM; // 8 x 16xPLD

  const int n    = blockIdx.y;
  const int row0 = blockIdx.x * 128;
  const int b    = row0 >> 11;
  const int tid  = threadIdx.x;
  const int lane = tid & 31, wave = tid >> 5;
  const int ln   = lane & 15, hf = lane >> 4;
  const int tq   = (row0 & (TT-1)) + wave*16;

  const unsigned short* kbase  = kp  + (size_t)b*TT*RR;
  const unsigned short* vtbase = vst + (((size_t)b*NSS + n)*RR)*TT;
  unsigned short* sp = s_p + wave*SPELEM;

  // async K-chunk copy: 32 rows x 512B into 528B-strided LDS rows
  auto issueK = [&](int s0, int buf){
    const unsigned dstb = (unsigned)(unsigned long long)(const void*)(s_kb + buf*SKELEM);
    const unsigned long long gb = (unsigned long long)(const void*)(kbase + (size_t)s0*RR);
#pragma unroll
    for (int j = 0; j < 4; ++j){
      const int i  = tid + j*256;
      const int s  = i >> 5, jj = i & 31;          // row s, 16B chunk jj
      const unsigned lofs = dstb + (unsigned)(s*SKLD*2 + jj*16);
      const unsigned long long ga = gb + (unsigned long long)i*16;
      asm volatile("global_load_async_to_lds_b128 %0, %1, off"
                   :: "v"(lofs), "v"(ga) : "memory");
    }
  };
  // async V-chunk copy: 256 rows x 64B (from transposed vst) into 80B-strided rows
  auto issueV = [&](int s0, int buf){
    const unsigned dstb = (unsigned)(unsigned long long)(const void*)(s_vtb + buf*SVELEM);
    const unsigned long long gb = (unsigned long long)(const void*)(vtbase + s0);
#pragma unroll
    for (int j = 0; j < 4; ++j){
      const int i  = tid + j*256;
      const int r  = i >> 2, jj = i & 3;           // row r, 16B chunk jj
      const unsigned lofs = dstb + (unsigned)(r*VLD*2 + jj*16);
      const unsigned long long ga = gb + (unsigned long long)r*(TT*2) + (unsigned long long)jj*16;
      asm volatile("global_load_async_to_lds_b128 %0, %1, off"
                   :: "v"(lofs), "v"(ga) : "memory");
    }
  };

  AFrag aq[8];
  {
    const unsigned short* qrow = qp + ((size_t)b*TT + tq)*RR;
#pragma unroll
    for (int kk = 0; kk < 8; ++kk) aq[kk] = load_frag_bf16(qrow + kk*32, RR, lane);
  }
  v8f o[16];
#pragma unroll
  for (int t = 0; t < 16; ++t)
#pragma unroll
    for (int e = 0; e < 8; ++e) o[t][e] = 0.f;
  float mrow[8], lrow[8];
#pragma unroll
  for (int e = 0; e < 8; ++e){ mrow[e] = -3.0e38f; lrow[e] = 0.f; }

  const float scale = 0.0625f;                   // 1/sqrt(256)
  const int send = (row0 & (TT-1)) + 128;        // causal upper bound (exclusive)

  // prologue: stage chunk 0
  issueK(0, 0);
  issueV(0, 0);
  asm volatile("s_wait_asynccnt 0x0" ::: "memory");
  __syncthreads();

  int cur = 0;
  for (int s0 = 0; s0 < send; s0 += 32){
    const int nxt = cur ^ 1;
    const bool more = (s0 + 32) < send;          // uniform across block
    if (more){ issueK(s0 + 32, nxt); issueV(s0 + 32, nxt); }

    const unsigned short* s_k  = s_kb  + cur*SKELEM;
    const unsigned short* s_vt = s_vtb + cur*SVELEM;

    // S = Q . K^T : B[k=r][n=s] = kp[s][r]
    v8f c0, c1;
#pragma unroll
    for (int e = 0; e < 8; ++e){ c0[e] = 0.f; c1[e] = 0.f; }
#pragma unroll
    for (int kk = 0; kk < 8; ++kk){
      AFrag b0 = load_frag_bf16(s_k + kk*32, SKLD, lane);
      AFrag b1 = load_frag_bf16(s_k + 16*SKLD + kk*32, SKLD, lane);
      c0 = wmma_bf16(aq[kk], b0, c0);
      c1 = wmma_bf16(aq[kk], b1, c1);
    }
    // mask (tril + score==0 -> -10000 quirk) and online softmax
#pragma unroll
    for (int e = 0; e < 8; ++e){
      const int trow = tq + e + 8*hf;
      float x0 = c0[e]*scale;
      float x1 = c1[e]*scale;
      if ((s0 + ln > trow)      || x0 == 0.f) x0 = -10000.f;
      if ((s0 + 16 + ln > trow) || x1 == 0.f) x1 = -10000.f;
      float mx = fmaxf(x0, x1);
      mx = fmaxf(mx, __shfl_xor(mx, 1, 32));
      mx = fmaxf(mx, __shfl_xor(mx, 2, 32));
      mx = fmaxf(mx, __shfl_xor(mx, 4, 32));
      mx = fmaxf(mx, __shfl_xor(mx, 8, 32));
      const float mnew = fmaxf(mrow[e], mx);
      const float p0 = __expf(x0 - mnew);
      const float p1 = __expf(x1 - mnew);
      const float fac = __expf(mrow[e] - mnew);
      mrow[e] = mnew;
      float ps = p0 + p1;
      ps += __shfl_xor(ps, 1, 32); ps += __shfl_xor(ps, 2, 32);
      ps += __shfl_xor(ps, 4, 32); ps += __shfl_xor(ps, 8, 32);
      lrow[e] = lrow[e]*fac + ps;
#pragma unroll
      for (int t = 0; t < 16; ++t) o[t][e] *= fac;
      sp[(e + 8*hf)*PLD + ln]      = bf(p0);
      sp[(e + 8*hf)*PLD + 16 + ln] = bf(p1);
    }
    // P is written and read by the same wave only; LDS is in-order per wave.
    AFrag ap = load_frag_bf16(sp, PLD, lane);
#pragma unroll
    for (int t = 0; t < 16; ++t){
      AFrag bv_ = load_frag_bf16(s_vt + (t*16)*VLD, VLD, lane); // B[k=s][n=r]
      o[t] = wmma_bf16(ap, bv_, o[t]);
    }

    asm volatile("s_wait_asynccnt 0x0" ::: "memory");
    __syncthreads();
    cur = nxt;
  }
  // normalize and store [B,NS,T,R]
#pragma unroll
  for (int e = 0; e < 8; ++e){
    const float inv = 1.0f / lrow[e];
    const size_t obase = (((size_t)b*NSS + n)*TT + tq + e + 8*hf)*RR;
#pragma unroll
    for (int t = 0; t < 16; ++t)
      out[obase + t*16 + ln] = o[t][e]*inv;
  }
}

// ---------------------------------------------------------------------------
// Kernel 3/4: covariance loss over proj [B,NS,T]
// ---------------------------------------------------------------------------
__global__ __launch_bounds__(256) void cov_kernel(const float* __restrict__ proj,
                                                  float* __restrict__ partial)
{
  const int b = blockIdx.x, tid = threadIdx.x;
  __shared__ float red[256];
  __shared__ float meanv[NSS];
  const float* P = proj + (size_t)b*NSS*TT;
  float s[NSS];
  for (int n = 0; n < NSS; ++n) s[n] = 0.f;
  for (int t = tid; t < TT; t += 256)
    for (int n = 0; n < NSS; ++n) s[n] += P[n*TT + t];
  for (int n = 0; n < NSS; ++n){
    red[tid] = s[n]; __syncthreads();
    for (int st = 128; st > 0; st >>= 1){ if (tid < st) red[tid] += red[tid+st]; __syncthreads(); }
    if (tid == 0) meanv[n] = red[0] / (float)TT;
    __syncthreads();
  }
  float c[NSS*NSS];
  for (int i = 0; i < NSS*NSS; ++i) c[i] = 0.f;
  for (int t = tid; t < TT; t += 256){
    float xc[NSS];
    for (int n = 0; n < NSS; ++n) xc[n] = P[n*TT + t] - meanv[n];
    for (int n = 0; n < NSS; ++n)
      for (int m = 0; m < NSS; ++m) c[n*NSS+m] += xc[n]*xc[m];
  }
  float tot = 0.f;
  for (int i = 0; i < NSS*NSS; ++i){
    red[tid] = c[i]; __syncthreads();
    for (int st = 128; st > 0; st >>= 1){ if (tid < st) red[tid] += red[tid+st]; __syncthreads(); }
    if (tid == 0) tot += fabsf(red[0] / (float)(TT-1));
    __syncthreads();
  }
  if (tid == 0) partial[b] = 0.5f * tot;
}

__global__ void final_kernel(const float* __restrict__ partial, float* __restrict__ outp){
  if (threadIdx.x == 0){
    float s = 0.f;
    for (int b = 0; b < BB; ++b) s += partial[b];
    outp[0] = s / (float)BB;
  }
}

// ---------------------------------------------------------------------------
extern "C" void kernel_launch(void* const* d_in, const int* in_sizes, int n_in,
                              void* d_out, int out_size, void* d_ws, size_t ws_size,
                              hipStream_t stream)
{
  const float* q  = (const float*)d_in[0];
  const float* k  = (const float*)d_in[1];
  const float* v  = (const float*)d_in[2];
  const float* wq = (const float*)d_in[3];
  const float* bq = (const float*)d_in[4];
  const float* wk = (const float*)d_in[5];
  const float* bk = (const float*)d_in[6];
  const float* wv = (const float*)d_in[7];
  const float* bv = (const float*)d_in[8];
  const float* wp = (const float*)d_in[9];
  const float* bp = (const float*)d_in[10];
  float* out = (float*)d_out;

  const size_t NQ = (size_t)BB*TT*RR;           // 8,388,608
  unsigned short* qbf = (unsigned short*)d_ws;  // bf16 copies of inputs
  unsigned short* kbf = qbf + NQ;
  unsigned short* vbf = kbf + NQ;
  unsigned short* wqb = vbf + NQ;               // [R,R]
  unsigned short* wkb = wqb + (size_t)RR*RR;
  unsigned short* wvb = wkb + (size_t)RR*RR;    // [NS,R,R]
  unsigned short* qp  = wvb + (size_t)NSS*RR*RR;
  unsigned short* kp  = qp + NQ;
  unsigned short* vst = kp + NQ;                // [B,NS,R,T]  (transposed V proj)
  float* proj    = (float*)(vst + (size_t)NSS*NQ);
  float* partial = proj + (size_t)BB*NSS*TT;

  cvt_kernel<<<2048, 256, 0, stream>>>(q, qbf, (int)(NQ/4));
  cvt_kernel<<<2048, 256, 0, stream>>>(k, kbf, (int)(NQ/4));
  cvt_kernel<<<2048, 256, 0, stream>>>(v, vbf, (int)(NQ/4));
  cvt_kernel<<<64,   256, 0, stream>>>(wq, wqb, RR*RR/4);
  cvt_kernel<<<64,   256, 0, stream>>>(wk, wkb, RR*RR/4);
  cvt_kernel<<<256,  256, 0, stream>>>(wv, wvb, NSS*RR*RR/4);

  const int shW = RR*SKLD * (int)sizeof(unsigned short);   // 132 KB
  dim3 gp(BB*TT/128, 2 + NSS);
  proj_kernel<<<gp, 256, shW, stream>>>(qbf,kbf,vbf,wqb,bq,wkb,bk,wvb,bv,wp,bp,qp,kp,vst,proj);

  const int shA = (2*SKELEM + 2*SVELEM + 8*SPELEM) * (int)sizeof(unsigned short);
  dim3 ga(BB*TT/128, NSS);
  attn_kernel<<<ga, 256, shA, stream>>>(qp, kp, vst, out);

  cov_kernel<<<BB, 256, 0, stream>>>(proj, partial);
  final_kernel<<<1, 32, 0, stream>>>(partial, out + (size_t)BB*NSS*TT*RR);
}
